// SDTransformerLayer_60043642798801
// MI455X (gfx1250) — matrix operand order repaired and checked by reading, not compiled, not run on hardware
//
#include <hip/hip_runtime.h>
#include <hip/hip_bf16.h>
#include <math.h>

// ---------------- types / WMMA helpers (gfx1250 wave32) ----------------
typedef __attribute__((ext_vector_type(16))) __bf16          v16bf;
typedef __attribute__((ext_vector_type(16))) unsigned short  v16us;
typedef __attribute__((ext_vector_type(8)))  float           v8f;

union BF16Frag { v16us u; v16bf b; };

__device__ __forceinline__ unsigned short f2bf(float f) {
  unsigned int u = __float_as_uint(f);
  u += 0x7FFFu + ((u >> 16) & 1u);          // round-to-nearest-even
  return (unsigned short)(u >> 16);
}

// hardware exp2: single v_exp_f32
__device__ __forceinline__ float hexp2(float x) { return __builtin_amdgcn_exp2f(x); }

// A-fragment: 16x32 bf16, row-major source with leading dim `ld` (elements).
// ISA layout: lane(0-15) row=lane, elems0-7 = K[half*8 .. +7], elems8-15 = K[half*8+16 .. +23]
__device__ __forceinline__ v16bf load_a_frag(const unsigned short* base, int row0,
                                             int k0, int ld, int lane) {
  const int m = lane & 15, half = lane >> 4;
  const unsigned short* p = base + (size_t)(row0 + m) * ld + k0 + half * 8;
  BF16Frag r;
#pragma unroll
  for (int e = 0; e < 8; ++e) r.u[e] = p[e];
#pragma unroll
  for (int e = 0; e < 8; ++e) r.u[8 + e] = p[16 + e];
  return r.b;
}

// B-fragment: 32x16 bf16 where source is stored B^T row-major: src[n*ld + k].
// ISA layout: lane holds col n=lane&15; elems0-7 = K[half*8 .. +7], elems8-15 = K[half*8+16 .. +23]
__device__ __forceinline__ v16bf load_bT_frag(const unsigned short* base, int n0,
                                              int k0, int ld, int lane) {
  const int n = lane & 15, half = lane >> 4;
  const unsigned short* p = base + (size_t)(n0 + n) * ld + k0 + half * 8;
  BF16Frag r;
#pragma unroll
  for (int e = 0; e < 8; ++e) r.u[e] = p[e];
#pragma unroll
  for (int e = 0; e < 8; ++e) r.u[8 + e] = p[16 + e];
  return r.b;
}

__device__ __forceinline__ v8f wmma_bf16(v16bf a, v16bf b, v8f c) {
  return __builtin_amdgcn_wmma_f32_16x16x32_bf16(false, a, false, b, (short)0, c,
                                                 false, false);
}

// ---------------- problem constants ----------------
#define NTOK   4096
#define DIM    192
#define HEADS  6
#define HD     32
#define HID    384      // DIM*2
#define CH     768      // 2*HID
#define HH     64
#define WW     64
#define QKDIM  576      // 3*DIM
#define LOG2E  1.4426950408889634f
// softmax scale folded with log2e so the inner loop uses raw v_exp (exp2)
#define SCALE_L2E (0.17677669529663687f * 1.4426950408889634f)

// ---------------- workspace layout (bytes) ----------------
#define OFF_XN1    (size_t)0                                  // [4096][192] bf16
#define OFF_Q      (OFF_XN1   + (size_t)NTOK*DIM*2)           // [6][4096][32] bf16 (pre-scaled)
#define OFF_K      (OFF_Q     + (size_t)HEADS*NTOK*HD*2)      // [6][4096][32] bf16
#define OFF_VT     (OFF_K     + (size_t)HEADS*NTOK*HD*2)      // [6][32][4096] bf16
#define OFF_ATTN   (OFF_VT    + (size_t)HEADS*HD*NTOK*2)      // [4096][192] bf16
#define OFF_X1     (OFF_ATTN  + (size_t)NTOK*DIM*2)           // [4096][192] f32
#define OFF_XN2    (OFF_X1    + (size_t)NTOK*DIM*4)           // [4096][192] bf16
#define OFF_T      (OFF_XN2   + (size_t)NTOK*DIM*2)           // [768][4096] f32
#define OFF_G      (OFF_T     + (size_t)CH*NTOK*4)            // [4096][384] bf16
#define OFF_WQKVT  (OFF_G     + (size_t)NTOK*HID*2)           // [576][192] bf16
#define OFF_WPROJT (OFF_WQKVT + (size_t)QKDIM*DIM*2)          // [192][192] bf16
#define OFF_WPIN   (OFF_WPROJT+ (size_t)DIM*DIM*2)            // [768][192] bf16
#define OFF_WPOUT  (OFF_WPIN  + (size_t)CH*DIM*2)             // [192][384] bf16

// ---------------- weight prep: f32 -> bf16 (with transposes for B^T loads) ----
__global__ void prep_weights_kernel(const float* __restrict__ wqkv,
                                    const float* __restrict__ wproj,
                                    const float* __restrict__ pinw,
                                    const float* __restrict__ poutw,
                                    unsigned short* wqkvT, unsigned short* wprojT,
                                    unsigned short* wpin, unsigned short* wpout) {
  int i = blockIdx.x * blockDim.x + threadIdx.x;
  const int N1 = QKDIM * DIM;       // 110592
  const int N2 = DIM * DIM;         // 36864
  const int N3 = CH * DIM;          // 147456
  const int N4 = DIM * HID;         // 73728
  if (i < N1) { int c = i / DIM, k = i % DIM; wqkvT[i] = f2bf(wqkv[k * QKDIM + c]); return; }
  i -= N1;
  if (i < N2) { int c = i / DIM, k = i % DIM; wprojT[i] = f2bf(wproj[k * DIM + c]); return; }
  i -= N2;
  if (i < N3) { wpin[i] = f2bf(pinw[i]); return; }   // pin_w is [O=768][I=192] already B^T
  i -= N3;
  if (i < N4) { wpout[i] = f2bf(poutw[i]); return; } // pout_w is [O=192][I=384] already B^T
}

// ---------------- LayerNorm: one wave per row of 192, output bf16 ------------
__global__ void ln_bf16_kernel(const float* __restrict__ x, const float* __restrict__ g,
                               const float* __restrict__ b, unsigned short* __restrict__ out) {
  int wave = (blockIdx.x * blockDim.x + threadIdx.x) >> 5;
  int lane = threadIdx.x & 31;
  if (wave >= NTOK) return;
  const float* xr = x + (size_t)wave * DIM;
  float v[6];
  float s = 0.f;
#pragma unroll
  for (int i = 0; i < 6; ++i) { v[i] = xr[lane + 32 * i]; s += v[i]; }
#pragma unroll
  for (int off = 1; off < 32; off <<= 1) s += __shfl_xor(s, off, 32);
  float mu = s * (1.f / DIM);
  float var = 0.f;
#pragma unroll
  for (int i = 0; i < 6; ++i) { float d = v[i] - mu; var += d * d; }
#pragma unroll
  for (int off = 1; off < 32; off <<= 1) var += __shfl_xor(var, off, 32);
  float rstd = rsqrtf(var * (1.f / DIM) + 1e-5f);
#pragma unroll
  for (int i = 0; i < 6; ++i) {
    int c = lane + 32 * i;
    out[(size_t)wave * DIM + c] = f2bf((v[i] - mu) * rstd * g[c] + b[c]);
  }
}

// ---------------- QKV GEMM (WMMA bf16), scatter into Q / K / V^T -------------
__global__ void qkv_kernel(const unsigned short* __restrict__ xn,
                           const unsigned short* __restrict__ wT,
                           const float* __restrict__ bqkv,
                           unsigned short* __restrict__ Qb,
                           unsigned short* __restrict__ Kb,
                           unsigned short* __restrict__ Vt) {
  int wid = (blockIdx.x * blockDim.x + threadIdx.x) >> 5;
  int lane = threadIdx.x & 31;
  int mt = wid / (QKDIM / 16), nt = wid % (QKDIM / 16);
  int r0 = mt * 16, c0 = nt * 16;
  v8f acc = {};
#pragma unroll
  for (int kk = 0; kk < DIM; kk += 32) {
    v16bf a = load_a_frag(xn, r0, kk, DIM, lane);
    v16bf b = load_bT_frag(wT, c0, kk, DIM, lane);
    acc = wmma_bf16(a, b, acc);
  }
  int half = lane >> 4, col = lane & 15;
  int c = c0 + col;
  float bias = bqkv[c];
#pragma unroll
  for (int v = 0; v < 8; ++v) {
    int r = r0 + v + half * 8;
    float val = acc[v] + bias;
    if (c < DIM) {                      // Q (fold softmax scale * log2e)
      int h = c >> 5, d = c & 31;
      Qb[((size_t)h * NTOK + r) * HD + d] = f2bf(val * SCALE_L2E);
    } else if (c < 2 * DIM) {           // K
      int c2 = c - DIM; int h = c2 >> 5, d = c2 & 31;
      Kb[((size_t)h * NTOK + r) * HD + d] = f2bf(val);
    } else {                            // V, stored transposed [dim][token]
      int c2 = c - 2 * DIM; int h = c2 >> 5, d = c2 & 31;
      Vt[((size_t)h * HD + d) * NTOK + r] = f2bf(val);
    }
  }
}

// ---- Flash attention with dense bias streamed non-temporally (the roofline) --
// Scores live in the exp2 domain: S2 = (q.k)*scale*log2e + bias*log2e, p=exp2(S2-m2).
__global__ __launch_bounds__(32)
void attn_kernel(const unsigned short* __restrict__ Qb,
                 const unsigned short* __restrict__ Kb,
                 const unsigned short* __restrict__ Vt,
                 const float* __restrict__ bias,
                 unsigned short* __restrict__ attn_out) {
  __shared__ unsigned short pL[16 * 32];          // P tile D-layout -> A-layout staging
  int lane = threadIdx.x & 31;
  int tile = blockIdx.x;                          // 6 heads * 256 q-blocks
  int h = tile >> 8;
  int q0 = (tile & 255) * 16;
  const unsigned short* Qh = Qb + (size_t)h * NTOK * HD;
  const unsigned short* Kh = Kb + (size_t)h * NTOK * HD;
  const unsigned short* Vh = Vt + (size_t)h * HD * NTOK;
  const float* biash = bias + (size_t)h * NTOK * NTOK + (size_t)q0 * NTOK;

  v16bf aq = load_a_frag(Qh, q0, 0, HD, lane);    // K=32 == head_dim: one fragment
  v8f acc0 = {}, acc1 = {};
  float mrow[8];    // running max (uniform across the 16-lane row group)
  float lpart[8];   // UNREDUCED per-lane partial of the softmax denominator
#pragma unroll
  for (int v = 0; v < 8; ++v) { mrow[v] = -INFINITY; lpart[v] = 0.f; }
  int half = lane >> 4, col = lane & 15;
  v8f zero = {};

  for (int j = 0; j < NTOK; j += 32) {
    // scores for 16 queries x 32 keys: two WMMAs
    v16bf kf0 = load_bT_frag(Kh, j,      0, HD, lane);
    v16bf kf1 = load_bT_frag(Kh, j + 16, 0, HD, lane);
    v8f s0 = wmma_bf16(aq, kf0, zero);
    v8f s1 = wmma_bf16(aq, kf1, zero);
    // add streamed bias * log2e (non-temporal: 402 MB read-once, larger than L2)
    const float* bp = biash + j + col;
#pragma unroll
    for (int v = 0; v < 8; ++v) {
      size_t r = (size_t)(v + half * 8) * NTOK;
      s0[v] = fmaf(__builtin_nontemporal_load(bp + r),      LOG2E, s0[v]);
      s1[v] = fmaf(__builtin_nontemporal_load(bp + r + 16), LOG2E, s1[v]);
    }
    // online softmax over the 32-key chunk (row = lanes of a 16-lane half)
#pragma unroll
    for (int v = 0; v < 8; ++v) {
      float cm = fmaxf(s0[v], s1[v]);
#pragma unroll
      for (int off = 1; off < 16; off <<= 1) cm = fmaxf(cm, __shfl_xor(cm, off, 32));
      float nm = fmaxf(mrow[v], cm);
      float alpha = hexp2(mrow[v] - nm);
      float p0 = hexp2(s0[v] - nm);
      float p1 = hexp2(s1[v] - nm);
      lpart[v] = lpart[v] * alpha + (p0 + p1);   // reduced only once, after the loop
      acc0[v] *= alpha;
      acc1[v] *= alpha;
      mrow[v] = nm;
      s0[v] = p0; s1[v] = p1;
    }
    // D-layout P -> LDS (bf16) -> A-layout fragment (per-wave DS ops are in-order)
#pragma unroll
    for (int v = 0; v < 8; ++v) {
      int r = v + half * 8;
      pL[r * 32 + col]      = f2bf(s0[v]);
      pL[r * 32 + col + 16] = f2bf(s1[v]);
    }
    __syncthreads();
    v16bf pf = load_a_frag(pL, 0, 0, 32, lane);
    __syncthreads();
    // accumulate O += P @ V  (V transposed => contiguous B-fragment loads)
    v16bf vf0 = load_bT_frag(Vh, 0,  j, NTOK, lane);
    v16bf vf1 = load_bT_frag(Vh, 16, j, NTOK, lane);
    acc0 = wmma_bf16(pf, vf0, acc0);
    acc1 = wmma_bf16(pf, vf1, acc1);
  }
  // finalize: reduce the deferred denominator, divide, write [4096][192] bf16
#pragma unroll
  for (int v = 0; v < 8; ++v) {
    float l = lpart[v];
#pragma unroll
    for (int off = 1; off < 16; off <<= 1) l += __shfl_xor(l, off, 32);
    int r = q0 + v + half * 8;
    float inv = 1.f / l;
    attn_out[(size_t)r * DIM + h * HD + col]      = f2bf(acc0[v] * inv);
    attn_out[(size_t)r * DIM + h * HD + 16 + col] = f2bf(acc1[v] * inv);
  }
}

// ---------------- proj GEMM + first residual -> x1 (f32) ---------------------
__global__ void proj_kernel(const unsigned short* __restrict__ attn,
                            const unsigned short* __restrict__ wT,
                            const float* __restrict__ bproj,
                            const float* __restrict__ x,
                            float* __restrict__ x1) {
  int wid = (blockIdx.x * blockDim.x + threadIdx.x) >> 5;
  int lane = threadIdx.x & 31;
  int mt = wid / (DIM / 16), nt = wid % (DIM / 16);
  int r0 = mt * 16, c0 = nt * 16;
  v8f acc = {};
#pragma unroll
  for (int kk = 0; kk < DIM; kk += 32) {
    v16bf a = load_a_frag(attn, r0, kk, DIM, lane);
    v16bf b = load_bT_frag(wT, c0, kk, DIM, lane);
    acc = wmma_bf16(a, b, acc);
  }
  int half = lane >> 4, col = lane & 15;
  int c = c0 + col;
  float bb = bproj[c];
#pragma unroll
  for (int v = 0; v < 8; ++v) {
    int r = r0 + v + half * 8;
    x1[(size_t)r * DIM + c] = acc[v] + bb + x[(size_t)r * DIM + c];
  }
}

// ---------------- pin 1x1 conv as GEMM -> t [768][4096] f32 ------------------
__global__ void pin_kernel(const unsigned short* __restrict__ xn2,
                           const unsigned short* __restrict__ wpin,
                           const float* __restrict__ pinb,
                           float* __restrict__ t) {
  int wid = (blockIdx.x * blockDim.x + threadIdx.x) >> 5;
  int lane = threadIdx.x & 31;
  int mt = wid / (CH / 16), nt = wid % (CH / 16);
  int r0 = mt * 16, c0 = nt * 16;
  v8f acc = {};
#pragma unroll
  for (int kk = 0; kk < DIM; kk += 32) {
    v16bf a = load_a_frag(xn2, r0, kk, DIM, lane);
    v16bf b = load_bT_frag(wpin, c0, kk, DIM, lane);
    acc = wmma_bf16(a, b, acc);
  }
  int half = lane >> 4, col = lane & 15;
  int o = c0 + col;
  float bb = pinb[o];
#pragma unroll
  for (int v = 0; v < 8; ++v) {
    int p = r0 + v + half * 8;                    // pixel index
    t[(size_t)o * NTOK + p] = acc[v] + bb;        // channel-major for depthwise conv
  }
}

// ---------------- depthwise 3x3 + GELU gate -> g [4096][384] bf16 ------------
__global__ void dwgelu_kernel(const float* __restrict__ t,
                              const float* __restrict__ dww,
                              const float* __restrict__ dwb,
                              unsigned short* __restrict__ g) {
  int i = blockIdx.x * blockDim.x + threadIdx.x;  // 384 * 4096
  if (i >= HID * NTOK) return;
  int c = i / NTOK, p = i % NTOK;
  int hh = p >> 6, ww = p & 63;
  float a1 = dwb[c], a2 = dwb[c + HID];
  const float* t1 = t + (size_t)c * NTOK;
  const float* t2 = t + (size_t)(c + HID) * NTOK;
  const float* w1 = dww + (size_t)c * 9;
  const float* w2 = dww + (size_t)(c + HID) * 9;
#pragma unroll
  for (int dy = -1; dy <= 1; ++dy) {
#pragma unroll
    for (int dx = -1; dx <= 1; ++dx) {
      int y = hh + dy, x = ww + dx;
      if (y >= 0 && y < HH && x >= 0 && x < WW) {
        int idx = y * WW + x;
        a1 += t1[idx] * w1[(dy + 1) * 3 + (dx + 1)];
        a2 += t2[idx] * w2[(dy + 1) * 3 + (dx + 1)];
      }
    }
  }
  float ge = 0.5f * a1 * (1.f + erff(a1 * 0.70710678118654752f)); // exact GELU
  g[(size_t)p * HID + c] = f2bf(ge * a2);
}

// ---------------- pout GEMM + second residual -> d_out -----------------------
__global__ void pout_kernel(const unsigned short* __restrict__ g,
                            const unsigned short* __restrict__ wpout,
                            const float* __restrict__ poutb,
                            const float* __restrict__ x1,
                            float* __restrict__ out) {
  int wid = (blockIdx.x * blockDim.x + threadIdx.x) >> 5;
  int lane = threadIdx.x & 31;
  int mt = wid / (DIM / 16), nt = wid % (DIM / 16);
  int r0 = mt * 16, c0 = nt * 16;
  v8f acc = {};
#pragma unroll
  for (int kk = 0; kk < HID; kk += 32) {
    v16bf a = load_a_frag(g, r0, kk, HID, lane);
    v16bf b = load_bT_frag(wpout, c0, kk, HID, lane);
    acc = wmma_bf16(a, b, acc);
  }
  int half = lane >> 4, col = lane & 15;
  int c = c0 + col;
  float bb = poutb[c];
#pragma unroll
  for (int v = 0; v < 8; ++v) {
    int r = r0 + v + half * 8;
    out[(size_t)r * DIM + c] = acc[v] + bb + x1[(size_t)r * DIM + c];
  }
}

// ---------------- launcher ----------------
extern "C" void kernel_launch(void* const* d_in, const int* in_sizes, int n_in,
                              void* d_out, int out_size, void* d_ws, size_t ws_size,
                              hipStream_t stream) {
  (void)in_sizes; (void)n_in; (void)out_size; (void)ws_size;
  const float* x     = (const float*)d_in[0];
  const float* sbias = (const float*)d_in[1];
  const float* ln1g  = (const float*)d_in[2];
  const float* ln1b  = (const float*)d_in[3];
  const float* wqkv  = (const float*)d_in[4];
  const float* bqkv  = (const float*)d_in[5];
  const float* wproj = (const float*)d_in[6];
  const float* bproj = (const float*)d_in[7];
  const float* ln2g  = (const float*)d_in[8];
  const float* ln2b  = (const float*)d_in[9];
  const float* pinw  = (const float*)d_in[10];
  const float* pinb  = (const float*)d_in[11];
  const float* dww   = (const float*)d_in[12];
  const float* dwb   = (const float*)d_in[13];
  const float* poutw = (const float*)d_in[14];
  const float* poutb = (const float*)d_in[15];
  float* out = (float*)d_out;

  char* ws = (char*)d_ws;
  unsigned short* xn1    = (unsigned short*)(ws + OFF_XN1);
  unsigned short* Qb     = (unsigned short*)(ws + OFF_Q);
  unsigned short* Kb     = (unsigned short*)(ws + OFF_K);
  unsigned short* Vt     = (unsigned short*)(ws + OFF_VT);
  unsigned short* attn   = (unsigned short*)(ws + OFF_ATTN);
  float*          x1     = (float*)        (ws + OFF_X1);
  unsigned short* xn2    = (unsigned short*)(ws + OFF_XN2);
  float*          tbuf   = (float*)        (ws + OFF_T);
  unsigned short* gbuf   = (unsigned short*)(ws + OFF_G);
  unsigned short* wqkvT  = (unsigned short*)(ws + OFF_WQKVT);
  unsigned short* wprojT = (unsigned short*)(ws + OFF_WPROJT);
  unsigned short* wpin   = (unsigned short*)(ws + OFF_WPIN);
  unsigned short* wpout  = (unsigned short*)(ws + OFF_WPOUT);

  // 1) weights -> bf16 (368640 elements)
  prep_weights_kernel<<<1440, 256, 0, stream>>>(wqkv, wproj, pinw, poutw,
                                                wqkvT, wprojT, wpin, wpout);
  // 2) LN1 (one wave/row, 8 waves/block)
  ln_bf16_kernel<<<NTOK / 8, 256, 0, stream>>>(x, ln1g, ln1b, xn1);
  // 3) QKV GEMM: 256 x 36 tiles, 8 waves/block
  qkv_kernel<<<(256 * (QKDIM / 16)) / 8, 256, 0, stream>>>(xn1, wqkvT, bqkv, Qb, Kb, Vt);
  // 4) flash attention w/ streamed bias: 6 heads * 256 q-blocks, 1 wave/block
  attn_kernel<<<HEADS * 256, 32, 0, stream>>>(Qb, Kb, Vt, sbias, attn);
  // 5) proj + residual: 256 x 12 tiles
  proj_kernel<<<(256 * (DIM / 16)) / 8, 256, 0, stream>>>(attn, wprojT, bproj, x, x1);
  // 6) LN2
  ln_bf16_kernel<<<NTOK / 8, 256, 0, stream>>>(x1, ln2g, ln2b, xn2);
  // 7) pin 1x1: 256 x 48 tiles
  pin_kernel<<<(256 * (CH / 16)) / 8, 256, 0, stream>>>(xn2, wpin, pinb, tbuf);
  // 8) depthwise 3x3 + gelu gate (1.57M threads)
  dwgelu_kernel<<<(HID * NTOK) / 256, 256, 0, stream>>>(tbuf, dww, dwb, gbuf);
  // 9) pout 1x1 + residual: 256 x 12 tiles
  pout_kernel<<<(256 * (DIM / 16)) / 8, 256, 0, stream>>>(gbuf, wpout, poutb, x1, out);
}
